// GPTBatchedSmall_53609781788780
// MI455X (gfx1250) — matrix operand order repaired and checked by reading, not compiled, not run on hardware
//
#include <hip/hip_runtime.h>
#include <math.h>

typedef __bf16 bf16;
typedef __attribute__((ext_vector_type(16))) __bf16 v16bf;
typedef __attribute__((ext_vector_type(8)))  float  v8f;
typedef __attribute__((ext_vector_type(4)))  unsigned int v4u;
typedef __attribute__((ext_vector_type(8)))  int v8i;
typedef __attribute__((ext_vector_type(4)))  int v4i;

#define EPSF 1.1920929e-07f
#define NB   4
#define TT   1024
#define ND   768
#define NH   6
#define NHD  128
#define NL   12
#define NBT  (NB*TT)
#define NVOC 50257

#if defined(__gfx1250__) && __has_builtin(__builtin_amdgcn_tensor_load_to_lds) && __has_builtin(__builtin_amdgcn_s_wait_tensorcnt)
#define USE_TDM 1
#else
#define USE_TDM 0
#endif

#if USE_TDM
// Issue one TDM load of a 2-D tile (data_size = 2 bytes / bf16) into LDS.
// Descriptor layout per CDNA5 ISA 8.3/8.4 (D# group0 128b, group1 256b; groups 2/3 zero => 2-D).
// clang-23 lane: 6-arg builtin (g0, g1, g2, g3, extra int32x8, cpol).
__device__ __forceinline__ void tdm_load_2d(unsigned lds_addr, const void* gptr,
                                            unsigned tensor_d0, unsigned tensor_d1,
                                            unsigned tile_d0, unsigned tile_d1,
                                            unsigned long long stride_d0_elems) {
  const unsigned long long ga = (unsigned long long)(size_t)gptr;
  v4u g0;
  g0.x = 1u;                                        // count=1 (valid), user mode
  g0.y = lds_addr;                                  // LDS byte address
  g0.z = (unsigned)(ga & 0xFFFFFFFFu);              // global_addr[31:0]
  g0.w = (unsigned)((ga >> 32) & 0x01FFFFFFu)       // global_addr[56:32]
       | (2u << 30);                                // type=2 ("image")
  v8i g1;
  g1[0] = (int)(1u << 16);                          // data_size=1 (2 bytes); mask/flags/pad = 0
  g1[1] = (int)((tensor_d0 & 0xFFFFu) << 16);       // atomic_barrier_addr=0 | tensor_dim0[15:0]
  g1[2] = (int)(((tensor_d0 >> 16) & 0xFFFFu) | ((tensor_d1 & 0xFFFFu) << 16));
  g1[3] = (int)(((tensor_d1 >> 16) & 0xFFFFu) | ((tile_d0 & 0xFFFFu) << 16));
  g1[4] = (int)(tile_d1 & 0xFFFFu);                 // tile_dim1 | tile_dim2=0
  g1[5] = (int)(stride_d0_elems & 0xFFFFFFFFu);     // tensor_dim0_stride[31:0]
  g1[6] = (int)((stride_d0_elems >> 32) & 0xFFFFu); // stride[47:32] | dim1_stride lo = 0
  g1[7] = 0;
  const v4i gz4 = {0, 0, 0, 0};
  const v8i gz8 = {0, 0, 0, 0, 0, 0, 0, 0};
  __builtin_amdgcn_tensor_load_to_lds(g0, g1, gz4, gz4, gz8, 0);
}
#endif

// ------------------------- reduction helpers -------------------------
__device__ __forceinline__ float warp_sum(float v) {
#pragma unroll
  for (int m = 1; m < 32; m <<= 1) v += __shfl_xor(v, m, 32);
  return v;
}
__device__ __forceinline__ float half_sum16(float v) {
#pragma unroll
  for (int m = 1; m < 16; m <<= 1) v += __shfl_xor(v, m, 32);
  return v;
}
__device__ __forceinline__ float half_max16(float v) {
#pragma unroll
  for (int m = 1; m < 16; m <<= 1) v = fmaxf(v, __shfl_xor(v, m, 32));
  return v;
}

// ------------------------- embedding gather + rmsnorm -------------------------
__global__ __launch_bounds__(256)
void k_embed_norm(const int* __restrict__ iseq, const float* __restrict__ embed,
                  float* __restrict__ x, float* __restrict__ x0) {
  const int row = blockIdx.x;
  const int tok = iseq[row];
  const float* e = embed + (size_t)tok * ND;
  __shared__ float red[8];
  float v0 = e[threadIdx.x], v1 = e[threadIdx.x + 256], v2 = e[threadIdx.x + 512];
  float ss = warp_sum(v0*v0 + v1*v1 + v2*v2);
  if ((threadIdx.x & 31) == 0) red[threadIdx.x >> 5] = ss;
  __syncthreads();
  float tot = 0.f;
#pragma unroll
  for (int i = 0; i < 8; ++i) tot += red[i];
  const float rn = rsqrtf(tot / (float)ND + EPSF);
  const size_t o = (size_t)row * ND + threadIdx.x;
  x[o]       = x0[o]       = v0 * rn;
  x[o + 256] = x0[o + 256] = v1 * rn;
  x[o + 512] = x0[o + 512] = v2 * rn;
}

// ------------------------- rmsnorm (f32 -> bf16) -------------------------
__global__ __launch_bounds__(256)
void k_rmsnorm_bf(const float* __restrict__ x, bf16* __restrict__ out) {
  const int row = blockIdx.x;
  const float* p = x + (size_t)row * ND;
  __shared__ float red[8];
  float v0 = p[threadIdx.x], v1 = p[threadIdx.x + 256], v2 = p[threadIdx.x + 512];
  float ss = warp_sum(v0*v0 + v1*v1 + v2*v2);
  if ((threadIdx.x & 31) == 0) red[threadIdx.x >> 5] = ss;
  __syncthreads();
  float tot = 0.f;
#pragma unroll
  for (int i = 0; i < 8; ++i) tot += red[i];
  const float rn = rsqrtf(tot / (float)ND + EPSF);
  bf16* o = out + (size_t)row * ND + threadIdx.x;
  o[0]   = (bf16)(v0 * rn);
  o[256] = (bf16)(v1 * rn);
  o[512] = (bf16)(v2 * rn);
}

// ------------------------- final rmsnorm (f32 -> f32) -------------------------
__global__ __launch_bounds__(256)
void k_norm_out(const float* __restrict__ x, float* __restrict__ out) {
  const int row = blockIdx.x;
  const float* p = x + (size_t)row * ND;
  __shared__ float red[8];
  float v0 = p[threadIdx.x], v1 = p[threadIdx.x + 256], v2 = p[threadIdx.x + 512];
  float ss = warp_sum(v0*v0 + v1*v1 + v2*v2);
  if ((threadIdx.x & 31) == 0) red[threadIdx.x >> 5] = ss;
  __syncthreads();
  float tot = 0.f;
#pragma unroll
  for (int i = 0; i < 8; ++i) tot += red[i];
  const float rn = rsqrtf(tot / (float)ND + EPSF);
  float* o = out + (size_t)row * ND + threadIdx.x;
  o[0]   = v0 * rn;
  o[256] = v1 * rn;
  o[512] = v2 * rn;
}

// ------------------------- residual / lambda mixing -------------------------
__global__ void k_mix(float* __restrict__ x, const float* __restrict__ x0,
                      const float* __restrict__ skip, const float* __restrict__ scal,
                      int layer, int n) {
  const int i = blockIdx.x * 256 + threadIdx.x;
  if (i >= n) return;
  const float l0 = scal[NL + 2*layer], l1 = scal[NL + 2*layer + 1];
  float v = x[i];
  if (skip) v += scal[layer - 6] * skip[i];
  x[i] = l0 * v + l1 * x0[i];
}

__global__ void k_copy(float* __restrict__ dst, const float* __restrict__ src, int n) {
  const int i = blockIdx.x * 256 + threadIdx.x;
  if (i < n) dst[i] = src[i];
}

// ------------------------- generic WMMA GEMM: C = A(bf16) @ W(f32->bf16)^T -------------------------
// A: [M x K] row-major bf16.  W: [N x K] row-major f32.
// relu2==0: Cf[o] = gemm (+ addsrc[o] if addsrc).  relu2==1: Cb[o] = bf16(relu(gemm)^2).
// Requires M,N % 128 == 0 and K % 32 == 0 (true for all call sites here).
__global__ __launch_bounds__(256)
void k_gemm(const bf16* __restrict__ A, const float* __restrict__ W,
            const float* __restrict__ addsrc, float* __restrict__ Cf,
            bf16* __restrict__ Cb, int M, int N, int K, int relu2) {
  __shared__ bf16 sA[128][32];
  __shared__ bf16 sB[128][32];
  const int tn0 = blockIdx.x * 128;
  const int tm0 = blockIdx.y * 128;
  const int tid  = threadIdx.x;
  const int lane = tid & 31;
  const int wave = tid >> 5;
  const int wm = (wave & 3) * 32;
  const int wn = (wave >> 2) * 64;
  const int mr    = lane & 15;
  const int kb8   = (lane >> 4) * 8;
  const int kb16  = (lane >> 4) * 16;
  const int lhalf = (lane >> 4) * 8;

  const v8f vzero = {0.f,0.f,0.f,0.f,0.f,0.f,0.f,0.f};
  v8f acc[2][4];
#pragma unroll
  for (int i = 0; i < 2; ++i)
#pragma unroll
    for (int j = 0; j < 4; ++j) acc[i][j] = vzero;

  const int ldrow = tid >> 1;
  const int ldcol = (tid & 1) * 16;

  for (int k0 = 0; k0 < K; k0 += 32) {
    __syncthreads();
#if USE_TDM
    // TDM: DMA the 128x32 bf16 A tile into LDS (wave 0 issues; TENSORcnt-tracked)
    if (wave == 0)
      tdm_load_2d((unsigned)(size_t)(void*)&sA[0][0],
                  A + (size_t)tm0 * K + k0,
                  (unsigned)K, (unsigned)M, 32u, 128u, (unsigned long long)K);
#else
    { // stage A tile (bf16, 16B vector loads)
      const bf16* srcA = A + (size_t)(tm0 + ldrow) * K + k0 + ldcol;
      *reinterpret_cast<uint4*>(&sA[ldrow][ldcol])     = *reinterpret_cast<const uint4*>(srcA);
      *reinterpret_cast<uint4*>(&sA[ldrow][ldcol + 8]) = *reinterpret_cast<const uint4*>(srcA + 8);
    }
#endif
    { // stage W tile (f32 -> bf16); TDM cannot convert, keep on VALU path
      const float* wrow = W + (size_t)(tn0 + ldrow) * K + k0 + ldcol;
      const float4* src = reinterpret_cast<const float4*>(wrow);
#pragma unroll
      for (int c = 0; c < 4; ++c) {
        float4 f = src[c];
        sB[ldrow][ldcol + 4*c + 0] = (bf16)f.x;
        sB[ldrow][ldcol + 4*c + 1] = (bf16)f.y;
        sB[ldrow][ldcol + 4*c + 2] = (bf16)f.z;
        sB[ldrow][ldcol + 4*c + 3] = (bf16)f.w;
      }
      if (k0 + 32 < K) __builtin_prefetch(wrow + 32, 0, 0); // next K-slice of weight stream
    }
#if USE_TDM
    if (wave == 0) __builtin_amdgcn_s_wait_tensorcnt(0);
#endif
    __syncthreads();

    v16bf af[2], bfr[4];
#pragma unroll
    for (int t = 0; t < 2; ++t) {   // A frags: 16x32, lane=m, K split by lane half
      const bf16* p = &sA[wm + t*16 + mr][0];
#pragma unroll
      for (int i = 0; i < 8; ++i) { af[t][i] = p[kb16 + i]; af[t][8 + i] = p[kb16 + 8 + i]; }
    }
#pragma unroll
    for (int t = 0; t < 4; ++t) {   // B frags: 32x16, lane=n, K 0-15 / 16-31 by half
      const bf16* p = &sB[wn + t*16 + mr][0];
#pragma unroll
      for (int i = 0; i < 16; ++i) bfr[t][i] = p[kb16 + i];
    }
#pragma unroll
    for (int i = 0; i < 2; ++i)
#pragma unroll
      for (int j = 0; j < 4; ++j)
        acc[i][j] = __builtin_amdgcn_wmma_f32_16x16x32_bf16(
            false, af[i], false, bfr[j], (short)0, acc[i][j], false, false);
  }

#pragma unroll
  for (int i = 0; i < 2; ++i)
#pragma unroll
    for (int j = 0; j < 4; ++j) {
      const int col = tn0 + wn + j*16 + mr;
#pragma unroll
      for (int r = 0; r < 8; ++r) {
        const int row = tm0 + wm + i*16 + lhalf + r;
        const size_t o = (size_t)row * N + col;
        float v = acc[i][j][r];
        if (relu2) {
          float t = v > 0.f ? v : 0.f;
          Cb[o] = (bf16)(t * t);
        } else {
          if (addsrc) v += addsrc[o];
          Cf[o] = v;
        }
      }
    }
}

// ------------------------- q/k head-norm + RoPE, v/ve mix; -> [B,H,T,HD] bf16 -------------------------
__global__ __launch_bounds__(128)
void k_qkv_post(const float* __restrict__ qkv, const int* __restrict__ iseq,
                const float* __restrict__ ve_w, const float* __restrict__ scal,
                int layer, int veIdx,
                bf16* __restrict__ qb, bf16* __restrict__ kb, bf16* __restrict__ vb) {
  const int bt = blockIdx.x;
  const int h  = blockIdx.y;
  const int t  = bt & (TT - 1);
  const int b  = bt >> 10;
  const int d  = threadIdx.x;
  const float* row = qkv + (size_t)bt * (3 * NH * NHD);
  float q = row[h*NHD + d];
  float k = row[(NH + h)*NHD + d];
  float v = row[(2*NH + h)*NHD + d];

  __shared__ float red[2][4];
  float sq = warp_sum(q*q);
  float sk = warp_sum(k*k);
  const int lane = threadIdx.x & 31, wv = threadIdx.x >> 5;
  if (lane == 0) { red[0][wv] = sq; red[1][wv] = sk; }
  __syncthreads();
  const float tq = red[0][0] + red[0][1] + red[0][2] + red[0][3];
  const float tk = red[1][0] + red[1][1] + red[1][2] + red[1][3];
  q *= rsqrtf(tq / (float)NHD + EPSF);
  k *= rsqrtf(tk / (float)NHD + EPSF);

  __shared__ float sh[2][NHD];
  sh[0][d] = q; sh[1][d] = k;
  __syncthreads();

  const int j = (d < 64) ? d : (d - 64);
  float c = 1.f, s = 0.f;
  if (j < 32) { // freq = (1/1024)^(j/31); upper 32 freqs are zero -> c=1,s=0
    const float fr = __expf(-6.93147180559945f * ((float)j / 31.0f));
    const float th = (float)t * fr;
    c = cosf(th); s = sinf(th);
  }
  float qo, ko;
  if (d < 64) { qo =  sh[0][d]*c + sh[0][d+64]*s;  ko =  sh[1][d]*c + sh[1][d+64]*s; }
  else        { qo = -sh[0][j]*s + sh[0][d]*c;     ko = -sh[1][j]*s + sh[1][d]*c; }

  const float sa0 = scal[3*NL + 2*layer], sa1 = scal[3*NL + 2*layer + 1];
  float vv = sa0 * v;
  if (veIdx >= 0) {
    const int tok = iseq[bt];
    vv += sa1 * ve_w[((size_t)veIdx * NVOC + tok) * ND + h*NHD + d];
  }
  const size_t o = (((size_t)(b*NH + h)) * TT + t) * NHD + d;
  qb[o] = (bf16)qo; kb[o] = (bf16)ko; vb[o] = (bf16)vv;
}

// ------------------------- causal flash attention (WMMA QK^T and PV) -------------------------
// grid: (T/128, H, B); 8 waves; wave w owns 16 query rows. Key blocks of 64 staged in LDS.
__global__ __launch_bounds__(256)
void k_attn(const bf16* __restrict__ Qg, const bf16* __restrict__ Kg,
            const bf16* __restrict__ Vg, bf16* __restrict__ Y) {
  __shared__ bf16 sK[64][NHD];
  __shared__ bf16 sV[64][NHD];
  __shared__ bf16 sP[8][16][64];

  const int qblk = blockIdx.x;
  const int h = blockIdx.y;
  const int b = blockIdx.z;
  const int tid  = threadIdx.x;
  const int lane = tid & 31;
  const int wave = tid >> 5;
  const int mr    = lane & 15;
  const int kb8   = (lane >> 4) * 8;
  const int kb16  = (lane >> 4) * 16;
  const int lhalf = (lane >> 4) * 8;

  const size_t bh = ((size_t)(b*NH + h)) * TT * NHD;
  const bf16* Qp = Qg + bh;
  const bf16* Kp = Kg + bh;
  const bf16* Vp = Vg + bh;

  const int q0 = qblk * 128 + wave * 16;

  v16bf qf[4]; // Q 16x128 as 4 A-fragments, kept in registers
  {
    const bf16* qrow = Qp + (size_t)(q0 + mr) * NHD;
#pragma unroll
    for (int s = 0; s < 4; ++s)
#pragma unroll
      for (int i = 0; i < 8; ++i) {
        qf[s][i]     = qrow[s*32 + kb16 + i];
        qf[s][8 + i] = qrow[s*32 + kb16 + 8 + i];
      }
  }

  const v8f vzero = {0.f,0.f,0.f,0.f,0.f,0.f,0.f,0.f};
  v8f oacc[8];
#pragma unroll
  for (int jj = 0; jj < 8; ++jj) oacc[jj] = vzero;
  float rmax[8], rsum[8];
#pragma unroll
  for (int r = 0; r < 8; ++r) { rmax[r] = -INFINITY; rsum[r] = 0.f; }

  const int nkb = (qblk + 1) * 2;
  for (int kblk = 0; kblk < nkb; ++kblk) {
    const int k0 = kblk * 64;
    __syncthreads();
#if USE_TDM
    // TDM: each 64x128 bf16 block is a contiguous 16KB run of [T][HD]; DMA K and V, wait TENSORcnt.
    if (wave == 0) {
      tdm_load_2d((unsigned)(size_t)(void*)&sK[0][0], Kp + (size_t)k0 * NHD,
                  (unsigned)(TT*NHD), 1u, 8192u, 1u, (unsigned long long)(TT*NHD));
      tdm_load_2d((unsigned)(size_t)(void*)&sV[0][0], Vp + (size_t)k0 * NHD,
                  (unsigned)(TT*NHD), 1u, 8192u, 1u, (unsigned long long)(TT*NHD));
      __builtin_amdgcn_s_wait_tensorcnt(0);
    }
#else
#pragma unroll
    for (int i = 0; i < 4; ++i) { // stage K/V block 64x128 bf16 each
      const int e  = tid + i * 256;
      const int rr = e >> 4;
      const int cc = (e & 15) * 8;
      *reinterpret_cast<uint4*>(&sK[rr][cc]) =
          *reinterpret_cast<const uint4*>(Kp + (size_t)(k0 + rr)*NHD + cc);
      *reinterpret_cast<uint4*>(&sV[rr][cc]) =
          *reinterpret_cast<const uint4*>(Vp + (size_t)(k0 + rr)*NHD + cc);
    }
#endif
    __syncthreads();

    v8f sc[4]; // S tile 16x64
#pragma unroll
    for (int nt = 0; nt < 4; ++nt) {
      sc[nt] = vzero;
#pragma unroll
      for (int s = 0; s < 4; ++s) {
        v16bf bk; // B frag: B[dk][n] = K[key=nt*16+n][d=s*32+dk]
        const bf16* p = &sK[nt*16 + mr][s*32];
#pragma unroll
        for (int i = 0; i < 16; ++i) bk[i] = p[kb16 + i];
        sc[nt] = __builtin_amdgcn_wmma_f32_16x16x32_bf16(
            false, qf[s], false, bk, (short)0, sc[nt], false, false);
      }
    }

    // online softmax update (C layout: lane=n, vgpr r -> m = lhalf+r)
#pragma unroll
    for (int r = 0; r < 8; ++r) {
      const int qg = q0 + lhalf + r;
      float mloc = -INFINITY;
#pragma unroll
      for (int nt = 0; nt < 4; ++nt) {
        float s = sc[nt][r] * 0.12f;
        if (k0 + nt*16 + mr > qg) s = -INFINITY; // causal mask
        sc[nt][r] = s;
        mloc = fmaxf(mloc, s);
      }
      mloc = half_max16(mloc);
      const float mnew = fmaxf(rmax[r], mloc);
      const float corr = __expf(rmax[r] - mnew);
      float ps = 0.f;
#pragma unroll
      for (int nt = 0; nt < 4; ++nt) {
        const float p = __expf(sc[nt][r] - mnew);
        sc[nt][r] = p;
        ps += p;
      }
      ps = half_sum16(ps);
      rsum[r] = rsum[r] * corr + ps;
      rmax[r] = mnew;
#pragma unroll
      for (int jj = 0; jj < 8; ++jj) oacc[jj][r] *= corr;
    }

    // P (C layout) -> LDS -> A-fragment layout, then PV
#pragma unroll
    for (int nt = 0; nt < 4; ++nt)
#pragma unroll
      for (int r = 0; r < 8; ++r)
        sP[wave][lhalf + r][nt*16 + mr] = (bf16)sc[nt][r];

#pragma unroll
    for (int s = 0; s < 2; ++s) {
      v16bf pa;
      const bf16* pp = &sP[wave][mr][s*32];
#pragma unroll
      for (int i = 0; i < 8; ++i) { pa[i] = pp[kb16 + i]; pa[8 + i] = pp[kb16 + 8 + i]; }
#pragma unroll
      for (int jj = 0; jj < 8; ++jj) {
        v16bf vfr; // B frag: B[k][n] = V[key=s*32+k][hd=jj*16+n]
#pragma unroll
        for (int i = 0; i < 16; ++i) vfr[i] = sV[s*32 + kb16 + i][jj*16 + mr];
        oacc[jj] = __builtin_amdgcn_wmma_f32_16x16x32_bf16(
            false, pa, false, vfr, (short)0, oacc[jj], false, false);
      }
    }
  }

  // epilogue: normalize and write token-major [B,T,H*HD] bf16
#pragma unroll
  for (int r = 0; r < 8; ++r) {
    const float inv = 1.0f / rsum[r];
    const int qg = q0 + lhalf + r;
#pragma unroll
    for (int jj = 0; jj < 8; ++jj)
      Y[((size_t)(b*TT) + qg) * ND + h*NHD + jj*16 + mr] = (bf16)(oacc[jj][r] * inv);
  }
}

// ------------------------- orchestration -------------------------
extern "C" void kernel_launch(void* const* d_in, const int* in_sizes, int n_in,
                              void* d_out, int out_size, void* d_ws, size_t ws_size,
                              hipStream_t stream) {
  (void)in_sizes; (void)n_in; (void)out_size; (void)ws_size;
  const int*   iseq    = (const int*)d_in[0];
  const float* embed_w = (const float*)d_in[1];
  const float* ve_w    = (const float*)d_in[2];
  const float* qkv_w   = (const float*)d_in[3];
  const float* aproj_w = (const float*)d_in[4];
  const float* fc_w    = (const float*)d_in[5];
  const float* mproj_w = (const float*)d_in[6];
  const float* scal    = (const float*)d_in[7];

  char* base = (char*)d_ws;
  size_t off = 0;
  auto alloc = [&](size_t bytes) -> char* {
    char* p = base + off;
    off += (bytes + 255) & ~(size_t)255;
    return p;
  };
  const size_t XE = (size_t)NBT * ND; // 3,145,728 elements per activation tensor
  float* x     = (float*)alloc(XE * 4);
  float* x0    = (float*)alloc(XE * 4);
  float* skips = (float*)alloc(6 * XE * 4);
  bf16*  xn    = (bf16*)alloc(XE * 2);
  bf16*  ybf   = (bf16*)alloc(XE * 2);
  bf16*  qb    = (bf16*)alloc(XE * 2);
  bf16*  kb    = (bf16*)alloc(XE * 2);
  bf16*  vb    = (bf16*)alloc(XE * 2);
  float* qkv   = (float*)alloc((size_t)NBT * 2304 * 4);
  bf16*  hbf   = (bf16*)qkv; // union: qkv dead (consumed by k_qkv_post) before hbf is written

  const int nElem = (int)XE;
  const int gElem = (nElem + 255) / 256;

  k_embed_norm<<<NBT, 256, 0, stream>>>(iseq, embed_w, x, x0);

  for (int l = 0; l < NL; ++l) {
    const float* skipPtr = (l >= 6) ? (skips + (size_t)(11 - l) * XE) : nullptr; // LIFO pop
    k_mix<<<gElem, 256, 0, stream>>>(x, x0, skipPtr, scal, l, nElem);

    if (l != 7) {
      k_rmsnorm_bf<<<NBT, 256, 0, stream>>>(x, xn);
      k_gemm<<<dim3(2304/128, NBT/128), 256, 0, stream>>>(
          xn, qkv_w + (size_t)l * 2304 * ND, nullptr, qkv, nullptr, NBT, 2304, ND, 0);
      const int ve = (l < 3) ? l : ((l >= 9) ? (l - 9) : -1);
      k_qkv_post<<<dim3(NBT, NH), 128, 0, stream>>>(qkv, iseq, ve_w, scal, l, ve, qb, kb, vb);
      k_attn<<<dim3(TT/128, NH, NB), 256, 0, stream>>>(qb, kb, vb, ybf);
      k_gemm<<<dim3(ND/128, NBT/128), 256, 0, stream>>>(
          ybf, aproj_w + (size_t)l * ND * ND, x, x, nullptr, NBT, ND, ND, 0);
    }

    k_rmsnorm_bf<<<NBT, 256, 0, stream>>>(x, xn);
    k_gemm<<<dim3(3072/128, NBT/128), 256, 0, stream>>>(
        xn, fc_w + (size_t)l * 3072 * ND, nullptr, nullptr, hbf, NBT, 3072, ND, 1);
    k_gemm<<<dim3(ND/128, NBT/128), 256, 0, stream>>>(
        hbf, mproj_w + (size_t)l * ND * 3072, x, x, nullptr, NBT, ND, 3072, 0);

    if (l < 6) k_copy<<<gElem, 256, 0, stream>>>(skips + (size_t)l * XE, x, nElem);
  }

  k_norm_out<<<NBT, 256, 0, stream>>>(x, (float*)d_out);
}
